// ScaledDotProductAttentionLayer_43860206027392
// MI455X (gfx1250) — compile-verified, matrix operand order checked by reading, not run on hardware
//
#include <hip/hip_runtime.h>
#include <hip/hip_bf16.h>

// ---------------------------------------------------------------------------
// ScaledDotProductAttention for MI455X (gfx1250, wave32, WMMA bf16)
//   q = x_q @ Wq^T + bq ; k = ... ; v = ...          (bf16 intermediates in ws)
//   out = softmax(causal(q k^T / 8)) v               (flash-style streaming)
// ---------------------------------------------------------------------------

typedef __attribute__((ext_vector_type(16))) __bf16 v16bf;
typedef __attribute__((ext_vector_type(8)))  __bf16 v8bf;
typedef __attribute__((ext_vector_type(8)))  float  v8f;

constexpr int   NB   = 8;      // batch
constexpr int   SEQ  = 2048;   // sequence length
constexpr int   DM   = 1024;   // d_model
constexpr int   DH   = 64;     // d_k == d_v
constexpr float SCL  = 0.125f; // 1/sqrt(64)
constexpr float NEGB = -1.0e30f;

// Load a 16-element bf16 WMMA operand for one lane.
// Pattern (per ISA 16-bit A layout, assumed mirrored for B):
//   elements [0..7]  = p[ko .. ko+7]        (one b128)
//   elements [8..15] = p[16+ko .. 16+ko+7]  (one b128)
// where ko = 8*(lane>=16).
static __device__ __forceinline__ v16bf load_op_bf16(const __bf16* __restrict__ p, int ko) {
  v8bf lo = *(const v8bf*)(p + ko);
  v8bf hi = *(const v8bf*)(p + 16 + ko);
  v16bf r;
#pragma unroll
  for (int i = 0; i < 8; ++i) { r[i] = lo[i]; r[8 + i] = hi[i]; }
  return r;
}

// Same pattern, but source is fp32 (converted to bf16 on the fly).
static __device__ __forceinline__ v16bf load_op_f32(const float* __restrict__ p, int ko) {
  v16bf r;
#pragma unroll
  for (int i = 0; i < 8; ++i) {
    r[i]     = (__bf16)p[ko + i];
    r[8 + i] = (__bf16)p[16 + ko + i];
  }
  return r;
}

static __device__ __forceinline__ v8f wmma_bf16(v16bf a, v16bf b, v8f c) {
  return __builtin_amdgcn_wmma_f32_16x16x32_bf16(false, a, false, b, (short)0, c,
                                                 false, false);
}

// ---------------------------------------------------------------------------
// Kernel 1: fused QKV projection. One wave computes a 16x64 output tile for
// one of {q,k,v}. q/k stored row-major bf16 [B][S][64]; v stored TRANSPOSED
// bf16 [B][64][S] so the attention kernel's V^T loads are contiguous.
// ---------------------------------------------------------------------------
__global__ __launch_bounds__(128) void qkv_proj_wmma(
    const float* __restrict__ xq, const float* __restrict__ xk,
    const float* __restrict__ xv,
    const float* __restrict__ Wq, const float* __restrict__ bq,
    const float* __restrict__ Wk, const float* __restrict__ bk,
    const float* __restrict__ Wv, const float* __restrict__ bv,
    __bf16* __restrict__ qbuf, __bf16* __restrict__ kbuf,
    __bf16* __restrict__ vtb) {
  const int wave = blockIdx.x * (blockDim.x >> 5) + (threadIdx.x >> 5);
  const int lane = threadIdx.x & 31;
  const int kind = wave % 3;          // 0=q 1=k 2=v
  const int tile = wave / 3;          // 0 .. B*S/16-1
  const int b    = tile >> 7;         // S/16 = 128 tiles per batch
  const int s0   = (tile & 127) << 4;
  const int row  = lane & 15;
  const int ko   = (lane >> 4) << 3;  // 0 or 8

  const float* x    = (kind == 0) ? xq : (kind == 1) ? xk : xv;
  const float* W    = (kind == 0) ? Wq : (kind == 1) ? Wk : Wv;
  const float* bias = (kind == 0) ? bq : (kind == 1) ? bk : bv;

  const float* xrow = x + ((size_t)b * SEQ + s0 + row) * DM;

  v8f acc[4];
#pragma unroll
  for (int t = 0; t < 4; ++t)
#pragma unroll
    for (int j = 0; j < 8; ++j) acc[t][j] = 0.0f;

  for (int kb0 = 0; kb0 < DM; kb0 += 32) {
    const v16bf a = load_op_f32(xrow + kb0, ko);  // A: 16 rows of x, K=32
#pragma unroll
    for (int t = 0; t < 4; ++t) {                 // B: W^T columns n=16t+row
      const float* wrow = W + (size_t)(t * 16 + row) * DM + kb0;
      const v16bf bb = load_op_f32(wrow, ko);
      acc[t] = wmma_bf16(a, bb, acc[t]);
    }
  }

  // C layout: column n = 16t + (lane&15); row m = j + ko
  if (kind < 2) {
    __bf16* dst = (kind == 0) ? qbuf : kbuf;
#pragma unroll
    for (int t = 0; t < 4; ++t) {
      const int   n  = t * 16 + row;
      const float bi = bias[n];
#pragma unroll
      for (int j = 0; j < 8; ++j) {
        const int m = ko + j;
        dst[((size_t)b * SEQ + s0 + m) * DH + n] = (__bf16)(acc[t][j] + bi);
      }
    }
  } else {
#pragma unroll
    for (int t = 0; t < 4; ++t) {
      const int   n  = t * 16 + row;
      const float bi = bias[n];
      union { v8bf v; __bf16 h[8]; } u;
#pragma unroll
      for (int j = 0; j < 8; ++j) u.h[j] = (__bf16)(acc[t][j] + bi);
      *(v8bf*)(vtb + ((size_t)b * DH + n) * SEQ + s0 + ko) = u.v;  // 16B store
    }
  }
}

// ---------------------------------------------------------------------------
// Kernel 2: flash attention. One wave owns 16 query rows; streams 32 keys per
// step. Scores computed TRANSPOSED (S^T = K_tile x Q^T) so each lane owns one
// query column -> softmax needs only shfl_xor(16); exp(S^T) lands directly in
// the B-operand layout for out^T = V^T x P^T. Causal mask in registers.
// ---------------------------------------------------------------------------
__global__ __launch_bounds__(128) void flash_attn_wmma(
    const __bf16* __restrict__ qbuf, const __bf16* __restrict__ kbuf,
    const __bf16* __restrict__ vtb, float* __restrict__ out) {
  const int wave = blockIdx.x * (blockDim.x >> 5) + (threadIdx.x >> 5);
  const int lane = threadIdx.x & 31;
  const int b    = wave >> 7;
  const int q0   = (wave & 127) << 4;
  const int row  = lane & 15;
  const int ko   = (lane >> 4) << 3;
  const int qabs = q0 + row;

  // Q operand (B-matrix of S^T): lane = query column, d in WMMA K pattern.
  const __bf16* qrow = qbuf + ((size_t)b * SEQ + q0 + row) * DH;
  const v16bf Bq0 = load_op_bf16(qrow, ko);       // d = 0..31
  const v16bf Bq1 = load_op_bf16(qrow + 32, ko);  // d = 32..63

  v8f accT[4];
#pragma unroll
  for (int t = 0; t < 4; ++t)
#pragma unroll
    for (int j = 0; j < 8; ++j) accT[t][j] = 0.0f;

  float mrun = NEGB, lrun = 0.0f;

  // keys kb0..kb0+31; last needed key is q0+15 -> kb0 <= q0 covers it.
  for (int kb0 = 0; kb0 <= q0; kb0 += 32) {
    const __bf16* kr0 = kbuf + ((size_t)b * SEQ + kb0 + row) * DH;
    const __bf16* kr1 = kr0 + 16 * DH;
    __builtin_prefetch(kr0 + 32 * DH, 0, 0);  // next key tile -> global_prefetch

    v8f s0t, s1t;
#pragma unroll
    for (int j = 0; j < 8; ++j) { s0t[j] = 0.0f; s1t[j] = 0.0f; }

    v16bf Ak;
    Ak  = load_op_bf16(kr0, ko);      s0t = wmma_bf16(Ak, Bq0, s0t);
    Ak  = load_op_bf16(kr0 + 32, ko); s0t = wmma_bf16(Ak, Bq1, s0t);
    Ak  = load_op_bf16(kr1, ko);      s1t = wmma_bf16(Ak, Bq0, s1t);
    Ak  = load_op_bf16(kr1 + 32, ko); s1t = wmma_bf16(Ak, Bq1, s1t);

    // S^T C layout: element j of s0t <-> key n = kb0+ko+j ; s1t <-> +16.
    float sv[16];
    float smax = NEGB;
#pragma unroll
    for (int j = 0; j < 8; ++j) {
      const int n0 = kb0 + ko + j;
      const float a0 = (n0 <= qabs) ? s0t[j] * SCL : NEGB;
      sv[j] = a0; smax = fmaxf(smax, a0);
      const int n1 = kb0 + 16 + ko + j;
      const float a1 = (n1 <= qabs) ? s1t[j] * SCL : NEGB;
      sv[8 + j] = a1; smax = fmaxf(smax, a1);
    }
    smax = fmaxf(smax, __shfl_xor(smax, 16, 32));  // partner lane = other keys

    const float mnew = fmaxf(mrun, smax);
    const float corr = __expf(mrun - mnew);
    float psum = 0.0f;
    v16bf pb;  // exp(S^T) -> directly in B-operand layout for K=32
#pragma unroll
    for (int e = 0; e < 16; ++e) {
      const float p = __expf(sv[e] - mnew);
      psum += p;
      pb[e] = (__bf16)p;
    }
    psum += __shfl_xor(psum, 16, 32);
    lrun = lrun * corr + psum;
    mrun = mnew;

#pragma unroll
    for (int t = 0; t < 4; ++t)
#pragma unroll
      for (int j = 0; j < 8; ++j) accT[t][j] *= corr;

    // out^T = V^T x P^T : A-operand rows are v-dims from transposed V buffer.
#pragma unroll
    for (int t = 0; t < 4; ++t) {
      const __bf16* vr = vtb + ((size_t)b * DH + t * 16 + row) * SEQ + kb0;
      const v16bf Av = load_op_bf16(vr, ko);
      accT[t] = wmma_bf16(Av, pb, accT[t]);
    }
  }

  // out^T C layout: column = query m = lane&15 ; row = vdim = 16t + ko + j.
  const float inv = 1.0f / lrun;
  float* orow = out + ((size_t)b * SEQ + q0 + row) * DH;
#pragma unroll
  for (int t = 0; t < 4; ++t) {
    float4 o0 = make_float4(accT[t][0] * inv, accT[t][1] * inv,
                            accT[t][2] * inv, accT[t][3] * inv);
    float4 o1 = make_float4(accT[t][4] * inv, accT[t][5] * inv,
                            accT[t][6] * inv, accT[t][7] * inv);
    *(float4*)(orow + t * 16 + ko)     = o0;
    *(float4*)(orow + t * 16 + ko + 4) = o1;
  }
}

// ---------------------------------------------------------------------------
extern "C" void kernel_launch(void* const* d_in, const int* in_sizes, int n_in,
                              void* d_out, int out_size, void* d_ws, size_t ws_size,
                              hipStream_t stream) {
  (void)in_sizes; (void)n_in; (void)out_size; (void)ws_size;
  const float* xq = (const float*)d_in[0];
  const float* xk = (const float*)d_in[1];
  const float* xv = (const float*)d_in[2];
  const float* Wq = (const float*)d_in[3];
  const float* bq = (const float*)d_in[4];
  const float* Wk = (const float*)d_in[5];
  const float* bk = (const float*)d_in[6];
  const float* Wv = (const float*)d_in[7];
  const float* bv = (const float*)d_in[8];
  float* out = (float*)d_out;

  // workspace: q,k row-major bf16 + v transposed bf16 (2 MB each, 6 MB total)
  __bf16* qb  = (__bf16*)d_ws;
  __bf16* kb  = qb + (size_t)NB * SEQ * DH;
  __bf16* vtb = kb + (size_t)NB * SEQ * DH;

  const int wavesProj = NB * (SEQ / 16) * 3;  // 3072 waves, 4 per block
  qkv_proj_wmma<<<wavesProj / 4, 128, 0, stream>>>(xq, xk, xv, Wq, bq, Wk, bk,
                                                   Wv, bv, qb, kb, vtb);

  const int wavesAttn = NB * (SEQ / 16);      // 1024 waves, 4 per block
  flash_attn_wmma<<<wavesAttn / 4, 128, 0, stream>>>(qb, kb, vtb, out);
}
